// MultiSequenceCrossAttention_90615220011041
// MI455X (gfx1250) — matrix operand order repaired and checked by reading, not compiled, not run on hardware
//
#include <hip/hip_runtime.h>
#include <hip/hip_bf16.h>

// ---------------- problem constants (match reference) ----------------
#define BB   2
#define LQ_  2048
#define LKV_ 2048
#define HH   1024
#define NH_  16
#define HD_  64            // HH / NH_
#define MROWS (BB * LQ_)   // 4096

typedef __attribute__((ext_vector_type(16))) __bf16 v16bf;
typedef __attribute__((ext_vector_type(4)))  __bf16 v4bf;
typedef __attribute__((ext_vector_type(8)))  float  v8f;
typedef __attribute__((ext_vector_type(4)))  float  f32x4;

__device__ inline unsigned lds_byte_off(const void* p) {
  // LDS aperture maps addr[31:0] -> LDS byte address (ISA 10.2)
  return (unsigned)(size_t)p;
}
__device__ inline void async_b128(unsigned lds, unsigned long long gaddr) {
  asm volatile("global_load_async_to_lds_b128 %0, %1, off"
               :: "v"(lds), "v"(gaddr) : "memory");
}

// =====================================================================
// 1) Segment positions + valid totals from attention_mask_in_length rows
// =====================================================================
__global__ void pos_kernel(const int* __restrict__ lens,
                           float* __restrict__ pos,
                           int* __restrict__ tot, int L) {
  if (threadIdx.x != 0) return;
  const int b = blockIdx.x;
  const int* row = lens + (long)b * L;
  float* p = pos + (long)b * L;
  long t = 0;
  long sum = 0;
  for (int i = 0; i < L; ++i) {
    int len = row[i];
    sum += len;
    for (int j = 0; j < len && t < L; ++j) { p[t] = (float)j; ++t; }
    if (t >= L) { for (int k = i + 1; k < L; ++k) sum += row[k]; break; }
  }
  for (; t < L; ++t) p[t] = 0.f;
  tot[b] = (int)(sum < (long)L ? sum : (long)L);
}

// =====================================================================
// 1b) fp32 -> bf16 bulk convert (vectorized, one-time per tensor)
// =====================================================================
__global__ void cvt_bf16_kernel(const float* __restrict__ src,
                                __bf16* __restrict__ dst, int n4) {
  int i = blockIdx.x * blockDim.x + threadIdx.x;
  if (i >= n4) return;
  f32x4 f = ((const f32x4*)src)[i];
  v4bf o;
  o[0] = (__bf16)f[0]; o[1] = (__bf16)f[1];
  o[2] = (__bf16)f[2]; o[3] = (__bf16)f[3];
  ((v4bf*)dst)[i] = o;
}

// =====================================================================
// 2) bf16 GEMM with bias: C[M,N] = (A[M,K] @ W[N,K]^T + bias) * scale
//    Inner loop: double-buffered async-to-LDS b128 copies + WMMA.
//    Block tile 64x128, 8 waves, K-step 32. 3 async copies per k-step.
// =====================================================================
#define TM 64
#define TN 128
#define TK 32

__global__ __launch_bounds__(256) void gemm_bf16_kernel(
    const __bf16* __restrict__ A, const __bf16* __restrict__ W,
    const float* __restrict__ bias, float* __restrict__ C,
    int M, int N, int K, float scale) {
  __shared__ __bf16 lA[2][TM * TK];   // [m][k], 4KB each
  __shared__ __bf16 lB[2][TN * TK];   // [n][k], 8KB each
  const int tid  = threadIdx.x;
  const int wave = tid >> 5, lane = tid & 31;
  const int lm = lane & 15, hi = lane >> 4;
  const int m0 = blockIdx.y * TM;
  const int n0 = blockIdx.x * TN;
  const int wm = (wave & 3) * 16;   // 4 row blocks of 16
  const int wn = (wave >> 2) * 64;  // 2 col blocks of 64

  // staging decomposition: thread -> (row = tid>>2, 16B part = tid&3)
  const int srow = tid >> 2, spart = tid & 3;
  const unsigned long long gA0 =
      (unsigned long long)(size_t)(A + (long)(m0 + srow) * K + spart * 8);
  const unsigned long long gB0 =
      (unsigned long long)(size_t)(W + (long)(n0 + srow) * K + spart * 8);
  const unsigned long long gB1 =
      (unsigned long long)(size_t)(W + (long)(n0 + srow + 64) * K + spart * 8);
  const unsigned lAoff = srow * 64 + spart * 16;           // row stride 64B
  const unsigned lBoff0 = srow * 64 + spart * 16;
  const unsigned lBoff1 = (srow + 64) * 64 + spart * 16;

#define GEMM_ISSUE(buf, k0)                                                   \
  do {                                                                        \
    unsigned long long kb_ = (unsigned long long)((k0) * 2);                  \
    async_b128(lds_byte_off(&lA[(buf)][0]) + lAoff, gA0 + kb_);               \
    async_b128(lds_byte_off(&lB[(buf)][0]) + lBoff0, gB0 + kb_);              \
    async_b128(lds_byte_off(&lB[(buf)][0]) + lBoff1, gB1 + kb_);              \
  } while (0)

  v8f acc[4] = {};
  GEMM_ISSUE(0, 0);
  const int nIter = K / TK;
  for (int it = 0; it < nIter; ++it) {
    const int cur = it & 1;
    __syncthreads();  // buffer cur^1 fully consumed by all waves
    if (it + 1 < nIter) {
      GEMM_ISSUE(cur ^ 1, (it + 1) * TK);
      asm volatile("s_wait_asynccnt 0x3" ::: "memory");  // current 3 landed
    } else {
      asm volatile("s_wait_asynccnt 0x0" ::: "memory");
    }
    __syncthreads();  // all waves' copies for buffer cur landed

    // A fragment: 16x32 (MxK), lane lm = row, hi selects K half pattern
    v16bf a;
#pragma unroll
    for (int v = 0; v < 8; ++v) {
      int kb = ((v & 4) << 2) + ((v & 3) << 1) + hi * 8;
      a[2 * v]     = lA[cur][(wm + lm) * TK + kb];
      a[2 * v + 1] = lA[cur][(wm + lm) * TK + kb + 1];
    }
#pragma unroll
    for (int t = 0; t < 4; ++t) {
      // B fragment: 32x16 (KxN), lane lm = col, hi selects K half
      v16bf bfr;
      int col = wn + t * 16 + lm;
#pragma unroll
      for (int v = 0; v < 8; ++v) {
        int k = ((v & 4) << 2) + ((v & 3) << 1) + hi * 8;
        bfr[2 * v]     = lB[cur][col * TK + k];
        bfr[2 * v + 1] = lB[cur][col * TK + k + 1];
      }
      acc[t] = __builtin_amdgcn_wmma_f32_16x16x32_bf16(
          false, a, false, bfr, (short)0, acc[t], false, false);
    }
  }
#undef GEMM_ISSUE
#pragma unroll
  for (int t = 0; t < 4; ++t) {
    int gcol = n0 + wn + t * 16 + lm;
    float bv = bias[gcol];
#pragma unroll
    for (int r = 0; r < 8; ++r) {
      int grow = m0 + wm + r + 8 * hi;
      C[(long)grow * N + gcol] = (acc[t][r] + bv) * scale;
    }
  }
}

// =====================================================================
// 3) RoPE in place (fp32). x element at ((b*L+t)*rowStride + h*HD + d),
//    pairs (d, d+HD/2) rotated with angle pos * 10000^(-2d/HD).
// =====================================================================
__global__ void rope_kernel(float* __restrict__ x, const float* __restrict__ pos,
                            int L, int rowStride) {
  const int n = BB * L * NH_ * (HD_ / 2);
  int idx = blockIdx.x * blockDim.x + threadIdx.x;
  if (idx >= n) return;
  int d = idx & 31;
  int tmp = idx >> 5;
  int h = tmp & (NH_ - 1);
  tmp >>= 4;
  int t = tmp % L;
  int b = tmp / L;
  float p = pos[(long)b * L + t];
  float f = p * __expf(-(2.0f * (float)d / (float)HD_) * 9.2103403719761836f);
  float c = __cosf(f), s = __sinf(f);
  long base = ((long)b * L + t) * (long)rowStride + (long)h * HD_;
  float x1 = x[base + d];
  float x2 = x[base + d + HD_ / 2];
  x[base + d]            = x1 * c - x2 * s;
  x[base + d + HD_ / 2]  = x2 * c + x1 * s;
}

// =====================================================================
// 3b) Pack fp32 (B,L,rowStride) slice into head-major bf16 [b][h][t][d]
//     so attention tiles are contiguous (async-copy friendly).
// =====================================================================
__global__ void pack_bf16_kernel(const float* __restrict__ src,
                                 __bf16* __restrict__ dst,
                                 int L, int rowStride, int colOff) {
  int idx = blockIdx.x * blockDim.x + threadIdx.x;
  const int n = BB * NH_ * L * (HD_ / 4);
  if (idx >= n) return;
  int d4 = idx & (HD_ / 4 - 1);   // 16 groups of 4
  int tmp = idx >> 4;
  int t = tmp % L;
  tmp /= L;
  int h = tmp & (NH_ - 1);
  int b = tmp >> 4;
  const float* s = src + ((long)b * L + t) * rowStride + colOff + (long)h * HD_ + d4 * 4;
  f32x4 f = *(const f32x4*)s;
  v4bf o;
  o[0] = (__bf16)f[0]; o[1] = (__bf16)f[1];
  o[2] = (__bf16)f[2]; o[3] = (__bf16)f[3];
  *(v4bf*)(dst + (((long)(b * NH_ + h) * L + t) * HD_ + d4 * 4)) = o;
}

// =====================================================================
// 4) Flash attention. Grid: (LQ/128, B*NH). 8 waves/block; each wave
//    owns a 16-row Q tile. Double-buffered async-to-LDS K/V chunks.
//    Output written as bf16 (consumed by bf16 O-projection GEMM).
// =====================================================================
__global__ __launch_bounds__(256) void attn_kernel(
    const __bf16* __restrict__ Qb,  // [b][h][t][d] bf16, roped & scaled
    const __bf16* __restrict__ Kb,  // [b][h][t][d] bf16, roped
    const __bf16* __restrict__ Vb,  // [b][h][t][d] bf16
    __bf16* __restrict__ O,         // (B, LQ, H) bf16
    const int* __restrict__ totQ, const int* __restrict__ totK) {
  __shared__ __bf16 lK[2][64 * 64];     // [kv][d], 8KB each
  __shared__ __bf16 lV[2][64 * 64];     // [kv][d], 8KB each
  __shared__ __bf16 lQs[8 * 16 * 64];   // per-wave Q tile [m][d]
  __shared__ __bf16 lPs[8 * 16 * 64];   // per-wave P tile [m][kv]

  const int tid = threadIdx.x, wave = tid >> 5, lane = tid & 31;
  const int lm = lane & 15, hi = lane >> 4;
  const int b = blockIdx.y / NH_, h = blockIdx.y % NH_;
  const int q0 = blockIdx.x * 128 + wave * 16;
  const int tq = totQ[b], tk = totK[b];
  __bf16* lQ = lQs + wave * 1024;
  __bf16* lP = lPs + wave * 1024;

  const size_t headQ = ((size_t)(b * NH_ + h) * LQ_);
  const size_t headK = ((size_t)(b * NH_ + h) * LKV_);
  const unsigned long long gKbase =
      (unsigned long long)(size_t)(Kb + headK * HD_) + tid * 16;
  const unsigned long long gVbase =
      (unsigned long long)(size_t)(Vb + headK * HD_) + tid * 16;

  // ---- stage this wave's Q tile (16x64 bf16 = 2KB, contiguous) async ----
  {
    unsigned ldst = lds_byte_off(lQ) + lane * 16;
    unsigned long long gsrc =
        (unsigned long long)(size_t)(Qb + (headQ + q0) * HD_) + lane * 16;
#pragma unroll
    for (int i = 0; i < 4; ++i)
      async_b128(ldst + i * 512, gsrc + (unsigned long long)(i * 512));
    asm volatile("s_wait_asynccnt 0x0" ::: "memory");
  }
  v16bf aq[2];
#pragma unroll
  for (int kk = 0; kk < 2; ++kk)
#pragma unroll
    for (int v = 0; v < 8; ++v) {
      int kb = kk * 32 + ((v & 4) << 2) + ((v & 3) << 1) + hi * 8;
      aq[kk][2 * v]     = lQ[lm * 64 + kb];
      aq[kk][2 * v + 1] = lQ[lm * 64 + kb + 1];
    }

  v8f o[4] = {};
  float Mx[8], Ls[8];
#pragma unroll
  for (int r = 0; r < 8; ++r) { Mx[r] = -1e30f; Ls[r] = 0.f; }

#define KV_ISSUE(buf, kv0)                                                    \
  do {                                                                        \
    unsigned long long go_ = (unsigned long long)((kv0) * HD_ * 2);           \
    unsigned lk_ = lds_byte_off(&lK[(buf)][0]) + tid * 16;                    \
    unsigned lv_ = lds_byte_off(&lV[(buf)][0]) + tid * 16;                    \
    async_b128(lk_,         gKbase + go_);                                    \
    async_b128(lk_ + 4096u, gKbase + go_ + 4096ull);                          \
    async_b128(lv_,         gVbase + go_);                                    \
    async_b128(lv_ + 4096u, gVbase + go_ + 4096ull);                          \
  } while (0)

  KV_ISSUE(0, 0);
  const int nIter = LKV_ / 64;
  for (int it = 0; it < nIter; ++it) {
    const int kv0 = it * 64, cur = it & 1;
    __syncthreads();  // buffer cur^1 fully consumed by all waves
    if (it + 1 < nIter) {
      KV_ISSUE(cur ^ 1, kv0 + 64);
      asm volatile("s_wait_asynccnt 0x4" ::: "memory");  // current 4 landed
    } else {
      asm volatile("s_wait_asynccnt 0x0" ::: "memory");
    }
    __syncthreads();  // all waves' copies for buffer cur landed
    const __bf16* cK = &lK[cur][0];
    const __bf16* cV = &lV[cur][0];

    // S = Q K^T  (4 col-tiles of 16 kv, 2 chained WMMAs each over HD=64)
    v8f s[4];
#pragma unroll
    for (int t = 0; t < 4; ++t) {
      v16bf bk0, bk1;
#pragma unroll
      for (int v = 0; v < 8; ++v) {
        int k = ((v & 4) << 2) + ((v & 3) << 1) + hi * 8;
        const int rowb = (t * 16 + lm) * 64;
        bk0[2 * v]     = cK[rowb + k];
        bk0[2 * v + 1] = cK[rowb + k + 1];
        bk1[2 * v]     = cK[rowb + 32 + k];
        bk1[2 * v + 1] = cK[rowb + 32 + k + 1];
      }
      v8f z = {};
      z = __builtin_amdgcn_wmma_f32_16x16x32_bf16(false, aq[0], false, bk0,
                                                  (short)0, z, false, false);
      z = __builtin_amdgcn_wmma_f32_16x16x32_bf16(false, aq[1], false, bk1,
                                                  (short)0, z, false, false);
      s[t] = z;
    }
    // mask out-of-range kv columns
#pragma unroll
    for (int t = 0; t < 4; ++t)
      if (kv0 + t * 16 + lm >= tk) {
#pragma unroll
        for (int r = 0; r < 8; ++r) s[t][r] = -1e9f;
      }

    // online softmax: row r of lane = q row (r + 8*hi)
    float alpha[8], rsum[8];
#pragma unroll
    for (int r = 0; r < 8; ++r) {
      float v = fmaxf(fmaxf(s[0][r], s[1][r]), fmaxf(s[2][r], s[3][r]));
      v = fmaxf(v, __shfl_xor(v, 1, 32));
      v = fmaxf(v, __shfl_xor(v, 2, 32));
      v = fmaxf(v, __shfl_xor(v, 4, 32));
      v = fmaxf(v, __shfl_xor(v, 8, 32));
      float mNew = fmaxf(Mx[r], v);
      alpha[r] = __expf(Mx[r] - mNew);
      Mx[r] = mNew;
      rsum[r] = 0.f;
    }
#pragma unroll
    for (int t = 0; t < 4; ++t)
#pragma unroll
      for (int r = 0; r < 8; ++r) {
        float p = __expf(s[t][r] - Mx[r]);
        rsum[r] += p;
        lP[(r + 8 * hi) * 64 + t * 16 + lm] = (__bf16)p;
        o[t][r] *= alpha[r];
      }
#pragma unroll
    for (int r = 0; r < 8; ++r) {
      float v = rsum[r];
      v += __shfl_xor(v, 1, 32);
      v += __shfl_xor(v, 2, 32);
      v += __shfl_xor(v, 4, 32);
      v += __shfl_xor(v, 8, 32);
      Ls[r] = Ls[r] * alpha[r] + v;
    }

    // O += P @ V   (re-layout P from C/D layout to A layout via LDS)
    v16bf ap[2];
#pragma unroll
    for (int kk = 0; kk < 2; ++kk)
#pragma unroll
      for (int v = 0; v < 8; ++v) {
        int kb = kk * 32 + ((v & 4) << 2) + ((v & 3) << 1) + hi * 8;
        ap[kk][2 * v]     = lP[lm * 64 + kb];
        ap[kk][2 * v + 1] = lP[lm * 64 + kb + 1];
      }
#pragma unroll
    for (int t = 0; t < 4; ++t) {
      v16bf bv0, bv1;
#pragma unroll
      for (int v = 0; v < 8; ++v) {
        int k = ((v & 4) << 2) + ((v & 3) << 1) + hi * 8;
        bv0[2 * v]     = cV[(k)*64 + t * 16 + lm];
        bv0[2 * v + 1] = cV[(k + 1) * 64 + t * 16 + lm];
        bv1[2 * v]     = cV[(32 + k) * 64 + t * 16 + lm];
        bv1[2 * v + 1] = cV[(32 + k + 1) * 64 + t * 16 + lm];
      }
      o[t] = __builtin_amdgcn_wmma_f32_16x16x32_bf16(false, ap[0], false, bv0,
                                                     (short)0, o[t], false, false);
      o[t] = __builtin_amdgcn_wmma_f32_16x16x32_bf16(false, ap[1], false, bv1,
                                                     (short)0, o[t], false, false);
    }
  }
#undef KV_ISSUE

  // normalize + zero invalid q rows, write out (bf16)
#pragma unroll
  for (int r = 0; r < 8; ++r) {
    int qrow = q0 + r + 8 * hi;
    float inv = Ls[r] > 0.f ? 1.f / Ls[r] : 0.f;
    if (qrow >= tq) inv = 0.f;
#pragma unroll
    for (int t = 0; t < 4; ++t)
      O[((long)(b * LQ_ + qrow)) * HH + (long)h * HD_ + t * 16 + lm] =
          (__bf16)(o[t][r] * inv);
  }
}

// =====================================================================
// Launch
// =====================================================================
extern "C" void kernel_launch(void* const* d_in, const int* in_sizes, int n_in,
                              void* d_out, int out_size, void* d_ws, size_t ws_size,
                              hipStream_t stream) {
  const float* q_states  = (const float*)d_in[0];
  const float* kv_states = (const float*)d_in[1];
  const int*   amq       = (const int*)d_in[2];
  const int*   amk       = (const int*)d_in[3];
  const float* Wq        = (const float*)d_in[4];
  const float* bq        = (const float*)d_in[5];
  const float* Wkv       = (const float*)d_in[6];
  const float* bkv       = (const float*)d_in[7];
  const float* Wo        = (const float*)d_in[8];
  const float* bo        = (const float*)d_in[9];
  float* out = (float*)d_out;

  // ---------------- workspace layout ----------------
  float*  q_ws   = (float*)d_ws;                           // 4096*1024 f32
  float*  kv_ws  = q_ws + (size_t)MROWS * HH;              // 4096*2048 f32
  __bf16* qs_bf  = (__bf16*)(kv_ws + (size_t)MROWS * 2 * HH);  // 4096*1024
  __bf16* kvs_bf = qs_bf + (size_t)MROWS * HH;             // 4096*1024
  __bf16* wq_bf  = kvs_bf + (size_t)MROWS * HH;            // 1024*1024
  __bf16* wkv_bf = wq_bf + (size_t)HH * HH;                // 2048*1024
  __bf16* wo_bf  = wkv_bf + (size_t)2 * HH * HH;           // 1024*1024
  __bf16* o_bf   = wo_bf + (size_t)HH * HH;                // 4096*1024
  __bf16* qb     = o_bf + (size_t)MROWS * HH;              // B*NH*LQ*HD
  __bf16* kb     = qb + (size_t)BB * NH_ * LQ_ * HD_;      // B*NH*LKV*HD
  __bf16* vb     = kb + (size_t)BB * NH_ * LKV_ * HD_;     // B*NH*LKV*HD
  float*  pos_q  = (float*)(vb + (size_t)BB * NH_ * LKV_ * HD_);
  float*  pos_k  = pos_q + (size_t)BB * LQ_;
  int*    tot_q  = (int*)(pos_k + (size_t)BB * LKV_);
  int*    tot_k  = tot_q + BB;

  // 1) positions
  pos_kernel<<<BB, 32, 0, stream>>>(amq, pos_q, tot_q, LQ_);
  pos_kernel<<<BB, 32, 0, stream>>>(amk, pos_k, tot_k, LKV_);

  // 1b) one-time bf16 conversion of GEMM operands
  {
    int n4;
    n4 = MROWS * HH / 4;
    cvt_bf16_kernel<<<(n4 + 255) / 256, 256, 0, stream>>>(q_states, qs_bf, n4);
    cvt_bf16_kernel<<<(n4 + 255) / 256, 256, 0, stream>>>(kv_states, kvs_bf, n4);
    n4 = HH * HH / 4;
    cvt_bf16_kernel<<<(n4 + 255) / 256, 256, 0, stream>>>(Wq, wq_bf, n4);
    cvt_bf16_kernel<<<(n4 + 255) / 256, 256, 0, stream>>>(Wo, wo_bf, n4);
    n4 = 2 * HH * HH / 4;
    cvt_bf16_kernel<<<(n4 + 255) / 256, 256, 0, stream>>>(Wkv, wkv_bf, n4);
  }

  // 2) projections (async-staged WMMA GEMMs); Q gets HD^-0.5 = 0.125
  gemm_bf16_kernel<<<dim3(HH / TN, MROWS / TM), 256, 0, stream>>>(
      qs_bf, wq_bf, bq, q_ws, MROWS, HH, HH, 0.125f);
  gemm_bf16_kernel<<<dim3(2 * HH / TN, MROWS / TM), 256, 0, stream>>>(
      kvs_bf, wkv_bf, bkv, kv_ws, MROWS, 2 * HH, HH, 1.0f);

  // 3) RoPE on q and on k half of kv (fp32)
  {
    int n = BB * LQ_ * NH_ * (HD_ / 2);
    rope_kernel<<<(n + 255) / 256, 256, 0, stream>>>(q_ws, pos_q, LQ_, HH);
    int nk = BB * LKV_ * NH_ * (HD_ / 2);
    rope_kernel<<<(nk + 255) / 256, 256, 0, stream>>>(kv_ws, pos_k, LKV_, 2 * HH);
  }

  // 3b) pack to head-major bf16 for async-copy-friendly attention tiles
  {
    int n = BB * NH_ * LQ_ * (HD_ / 4);
    pack_bf16_kernel<<<(n + 255) / 256, 256, 0, stream>>>(q_ws, qb, LQ_, HH, 0);
    int nk = BB * NH_ * LKV_ * (HD_ / 4);
    pack_bf16_kernel<<<(nk + 255) / 256, 256, 0, stream>>>(kv_ws, kb, LKV_, 2 * HH, 0);
    pack_bf16_kernel<<<(nk + 255) / 256, 256, 0, stream>>>(kv_ws, vb, LKV_, 2 * HH, HH);
  }

  // 4) attention (double-buffered async-to-LDS, WMMA), bf16 output
  attn_kernel<<<dim3(LQ_ / 128, BB * NH_), 256, 0, stream>>>(
      qb, kb, vb, o_bf, tot_q, tot_k);

  // 5) output projection
  gemm_bf16_kernel<<<dim3(HH / TN, MROWS / TM), 256, 0, stream>>>(
      o_bf, wo_bf, bo, out, MROWS, HH, HH, 1.0f);
}
